// MsPAM_CAM_Layer_38233798869178
// MI455X (gfx1250) — matrix operand order repaired and checked by reading, not compiled, not run on hardware
//
#include <hip/hip_runtime.h>

typedef __attribute__((ext_vector_type(16))) _Float16 v16h;
typedef __attribute__((ext_vector_type(8)))  _Float16 v8h;
typedef __attribute__((ext_vector_type(8)))  float    v8f;

#define NN 4096
#define NB 2
#define NC 64
#define NT 256  // NN/16 column tiles

// ---------------- WMMA helpers (CDNA5 wave32 layouts) ------------------------
__device__ __forceinline__ v8f wmma16(v16h a, v16h b, v8f c) {
  return __builtin_amdgcn_wmma_f32_16x16x32_f16(false, a, false, b, (short)0, c,
                                                false, false);
}

// A-operand (16x32 f16) from a row-major matrix: lane needs row = lane%16 and
// K in {kb..kb+7} U {16+kb..16+kb+7} (kb = 0 / 8) -> two contiguous 8-f16 runs.
__device__ __forceinline__ v16h load_A_rm(const _Float16* A, size_t lda) {
  int lane = threadIdx.x & 31;
  int row = lane & 15;
  int kb = (lane < 16) ? 0 : 8;
  const _Float16* p = A + (size_t)row * lda + kb;
  v8h lo = *(const v8h*)p;
  v8h hi = *(const v8h*)(p + 16);
  return __builtin_shufflevector(lo, hi, 0, 1, 2, 3, 4, 5, 6, 7, 8, 9, 10, 11,
                                 12, 13, 14, 15);
}

// A-operand from the pre-permuted Q layout (row = [k0..7,k16..23,k8..15,k24..31]):
// one contiguous 32-byte load per lane.
__device__ __forceinline__ v16h load_A_qf(const _Float16* Qf, int m0) {
  int lane = threadIdx.x & 31;
  int row = lane & 15;
  int sel = (lane < 16) ? 0 : 16;
  return *(const v16h*)(Qf + (size_t)(m0 + row) * 32 + sel);
}

// B-operand (32x16 f16) when the tile is stored column-major (col stride ldb):
// element [k][col] at Bp[col*ldb + k] -> lane reads K {kb..kb+7} U {16+kb..16+kb+7}.
__device__ __forceinline__ v16h load_B_cm(const _Float16* Bp, size_t ldb) {
  int lane = threadIdx.x & 31;
  int col = lane & 15;
  int kb = (lane < 16) ? 0 : 8;
  const _Float16* p = Bp + (size_t)col * ldb + kb;
  v8h lo = *(const v8h*)p;
  v8h hi = *(const v8h*)(p + 16);
  return __builtin_shufflevector(lo, hi, 0, 1, 2, 3, 4, 5, 6, 7, 8, 9, 10, 11,
                                 12, 13, 14, 15);
}

// ---------------- Kernel 1: projections -> WMMA-native f16 workspace ---------
// Qf : [B][N][32]  permuted rows (one 32B frag chunk per lane half)
// Kf : [3][B][NT][32 lanes][16]  fragment-native B-operand tiles
// Vh : [B][64][N] row-major (A-operand via load_A_rm)
// xh/yh/zh : [B][64][N] f16 copies for CAM gram
__global__ void proj_kernel(const float* __restrict__ x, const float* __restrict__ y,
                            const float* __restrict__ z,
                            const float* __restrict__ wq, const float* __restrict__ bq,
                            const float* __restrict__ wk1, const float* __restrict__ bk1,
                            const float* __restrict__ wk2, const float* __restrict__ bk2,
                            const float* __restrict__ wk3, const float* __restrict__ bk3,
                            const float* __restrict__ wv, const float* __restrict__ bv,
                            _Float16* __restrict__ Qf, _Float16* __restrict__ Kf,
                            _Float16* __restrict__ Vh, _Float16* __restrict__ xh,
                            _Float16* __restrict__ yh, _Float16* __restrict__ zh) {
  __shared__ float sw[6144];  // wq(512) wk1(512) wk2(512) wk3(512) wv(4096)
  for (int i = threadIdx.x; i < 512; i += 256) sw[i] = wq[i];
  for (int i = threadIdx.x; i < 512; i += 256) sw[512 + i] = wk1[i];
  for (int i = threadIdx.x; i < 512; i += 256) sw[1024 + i] = wk2[i];
  for (int i = threadIdx.x; i < 512; i += 256) sw[1536 + i] = wk3[i];
  for (int i = threadIdx.x; i < 4096; i += 256) sw[2048 + i] = wv[i];
  __syncthreads();

  int t = blockIdx.x * 256 + threadIdx.x;  // B*N threads
  int b = t >> 12, n = t & (NN - 1);
  int tile = n >> 4, col = n & 15;

  float xv[NC];
#pragma unroll
  for (int c = 0; c < NC; ++c) {
    xv[c] = x[((size_t)b * NC + c) * NN + n];
    xh[((size_t)b * NC + c) * NN + n] = (_Float16)xv[c];
  }

  // Q: only k<8 nonzero; permuted row => values land in slots 0..7, rest zero.
  _Float16* Qrow = Qf + ((size_t)b * NN + n) * 32;
  for (int k = 0; k < 8; ++k) {
    float a = bq[k];
#pragma unroll
    for (int c = 0; c < NC; ++c) a += sw[k * 64 + c] * xv[c];
    Qrow[k] = (_Float16)a;
  }
#pragma unroll
  for (int k = 8; k < 32; ++k) Qrow[k] = (_Float16)0.f;

  // K1 from x -> fragment-native layout
  {
    _Float16* kfl = Kf + (((size_t)b * NT + tile) * 32 + col) * 16;
    for (int k = 0; k < 8; ++k) {
      float a = bk1[k];
#pragma unroll
      for (int c = 0; c < NC; ++c) a += sw[512 + k * 64 + c] * xv[c];
      kfl[k] = (_Float16)a;
    }
#pragma unroll
    for (int k = 8; k < 16; ++k) kfl[k] = (_Float16)0.f;
#pragma unroll
    for (int e = 0; e < 16; ++e) kfl[256 + e] = (_Float16)0.f;  // lane col+16
  }

  // V from x
  _Float16* Vp = Vh + (size_t)b * NC * NN;
  for (int o = 0; o < NC; ++o) {
    float a = bv[o];
#pragma unroll
    for (int c = 0; c < NC; ++c) a += sw[2048 + o * 64 + c] * xv[c];
    Vp[(size_t)o * NN + n] = (_Float16)a;
  }

  // K2 from y
#pragma unroll
  for (int c = 0; c < NC; ++c) {
    xv[c] = y[((size_t)b * NC + c) * NN + n];
    yh[((size_t)b * NC + c) * NN + n] = (_Float16)xv[c];
  }
  {
    _Float16* kfl = Kf + (((size_t)(NB + b) * NT + tile) * 32 + col) * 16;
    for (int k = 0; k < 8; ++k) {
      float a = bk2[k];
#pragma unroll
      for (int c = 0; c < NC; ++c) a += sw[1024 + k * 64 + c] * xv[c];
      kfl[k] = (_Float16)a;
    }
#pragma unroll
    for (int k = 8; k < 16; ++k) kfl[k] = (_Float16)0.f;
#pragma unroll
    for (int e = 0; e < 16; ++e) kfl[256 + e] = (_Float16)0.f;
  }

  // K3 from z
#pragma unroll
  for (int c = 0; c < NC; ++c) {
    xv[c] = z[((size_t)b * NC + c) * NN + n];
    zh[((size_t)b * NC + c) * NN + n] = (_Float16)xv[c];
  }
  {
    _Float16* kfl = Kf + (((size_t)(2 * NB + b) * NT + tile) * 32 + col) * 16;
    for (int k = 0; k < 8; ++k) {
      float a = bk3[k];
#pragma unroll
      for (int c = 0; c < NC; ++c) a += sw[1536 + k * 64 + c] * xv[c];
      kfl[k] = (_Float16)a;
    }
#pragma unroll
    for (int k = 8; k < 16; ++k) kfl[k] = (_Float16)0.f;
#pragma unroll
    for (int e = 0; e < 16; ++e) kfl[256 + e] = (_Float16)0.f;
  }
}

// ---------------- Kernel 2: PAM pass 1 — row max, then row sum-exp -----------
// grid (N/16/8, B, 3), block 256 (8 waves); wave owns rows m0..m0+15
__global__ void pam_pass1(const _Float16* __restrict__ Qf,
                          const _Float16* __restrict__ Kf,
                          float2* __restrict__ MR) {
  int wave = threadIdx.x >> 5, lane = threadIdx.x & 31;
  int m0 = (blockIdx.x * 8 + wave) * 16;
  int b = blockIdx.y, br = blockIdx.z;
  const _Float16* Qb = Qf + (size_t)b * NN * 32;
  const _Float16* Kfb = Kf + (size_t)(br * NB + b) * NT * 32 * 16;

  v16h a = load_A_qf(Qb, m0);

  float lm[8];
#pragma unroll
  for (int r = 0; r < 8; ++r) lm[r] = -1e30f;
  for (int t = 0; t < NT; ++t) {
    v16h bf = *(const v16h*)(Kfb + ((size_t)t * 32 + lane) * 16);
    v8f c = {};
    c = wmma16(a, bf, c);
#pragma unroll
    for (int r = 0; r < 8; ++r) lm[r] = fmaxf(lm[r], c[r]);
  }
#pragma unroll
  for (int off = 8; off >= 1; off >>= 1)
#pragma unroll
    for (int r = 0; r < 8; ++r) lm[r] = fmaxf(lm[r], __shfl_xor(lm[r], off));

  float ls[8];
#pragma unroll
  for (int r = 0; r < 8; ++r) ls[r] = 0.f;
  for (int t = 0; t < NT; ++t) {
    v16h bf = *(const v16h*)(Kfb + ((size_t)t * 32 + lane) * 16);
    v8f c = {};
    c = wmma16(a, bf, c);
#pragma unroll
    for (int r = 0; r < 8; ++r) ls[r] += __expf(c[r] - lm[r]);
  }
#pragma unroll
  for (int off = 8; off >= 1; off >>= 1)
#pragma unroll
    for (int r = 0; r < 8; ++r) ls[r] += __shfl_xor(ls[r], off);

  if ((lane & 15) == 0) {
    int half8 = (lane < 16) ? 0 : 8;
    size_t base = ((size_t)br * NB + b) * NN;
#pragma unroll
    for (int r = 0; r < 8; ++r)
      MR[base + m0 + r + half8] = make_float2(lm[r], 1.0f / ls[r]);
  }
}

// ---------------- Kernel 3: PAM pass 2 — out = V @ (sum_br P_br) -------------
// 2048 waves: wave owns 16x16 tile (b, c0, n0); 256 blocks x 256 threads
#define PSTRIDE 36
__global__ void pam_pass2(const _Float16* __restrict__ Qf,
                          const _Float16* __restrict__ Kf,
                          const _Float16* __restrict__ Vh,
                          const float2* __restrict__ MR,
                          float* __restrict__ pamacc) {
  __shared__ float sP[8][16 * PSTRIDE];
  int wslot = threadIdx.x >> 5, lane = threadIdx.x & 31;
  int wid = blockIdx.x * 8 + wslot;
  int n0 = (wid & 255) * 16;
  int c0 = ((wid >> 8) & 3) * 16;
  int b = wid >> 10;
  int half8 = (lane < 16) ? 0 : 8;
  int col = lane & 15;
  int kb16 = (lane < 16) ? 0 : 16;
  float* Pst = sP[wslot];

  const _Float16* Qb = Qf + (size_t)b * NN * 32;
  int tile = n0 >> 4;
  v16h bK[3];
#pragma unroll
  for (int br = 0; br < 3; ++br)
    bK[br] =
        *(const v16h*)(Kf + (((size_t)(br * NB + b) * NT + tile) * 32 + lane) * 16);

  v8f acc = {};
  for (int m0 = 0; m0 < NN; m0 += 32) {
    v16h aQ0 = load_A_qf(Qb, m0);
    v16h aQ1 = load_A_qf(Qb, m0 + 16);
    float psum[16];
#pragma unroll
    for (int i = 0; i < 16; ++i) psum[i] = 0.f;

#pragma unroll
    for (int br = 0; br < 3; ++br) {
      v8f z0 = {}, z1 = {};
      v8f e0 = wmma16(aQ0, bK[br], z0);
      v8f e1 = wmma16(aQ1, bK[br], z1);
      const float2* MRb = MR + ((size_t)br * NB + b) * NN;
#pragma unroll
      for (int r = 0; r < 8; ++r) {
        float2 s0 = MRb[m0 + r + half8];
        float2 s1 = MRb[m0 + 16 + r + half8];
        psum[r] += __expf(e0[r] - s0.x) * s0.y;
        psum[8 + r] += __expf(e1[r] - s1.x) * s1.y;
      }
    }
    // stage P transposed [col][m_local]: two contiguous 8-dword runs per lane
#pragma unroll
    for (int r = 0; r < 8; ++r) {
      Pst[col * PSTRIDE + half8 + r] = psum[r];
      Pst[col * PSTRIDE + 16 + half8 + r] = psum[8 + r];
    }
    v16h aV = load_A_rm(Vh + ((size_t)b * NC + c0) * NN + m0, NN);
    v16h bP;
#pragma unroll
    for (int e = 0; e < 16; ++e)
      bP[e] = (_Float16)Pst[col * PSTRIDE + (e < 8 ? half8 + e : 8 + half8 + e)];  // two 8-dword runs
    acc = wmma16(aV, bP, acc);
  }
  float* PA = pamacc + (size_t)b * NC * NN;
#pragma unroll
  for (int r = 0; r < 8; ++r)
    PA[(size_t)(c0 + r + half8) * NN + n0 + col] = acc[r];
}

// ---------------- Kernel 4: CAM gram matrices via WMMA -----------------------
// 96 waves (12 blocks x 8): wave owns one 16x16 tile of G[b][br]
__global__ void cam_gram(const _Float16* __restrict__ xh, const _Float16* __restrict__ yh,
                         const _Float16* __restrict__ zh, float* __restrict__ camG) {
  int wslot = threadIdx.x >> 5, lane = threadIdx.x & 31;
  int wid = blockIdx.x * 8 + wslot;
  int b = wid / 48;
  int rem = wid % 48;
  int br = rem >> 4;
  int t16 = rem & 15;
  int c0 = (t16 >> 2) * 16, d0 = (t16 & 3) * 16;
  const _Float16* xa = xh + ((size_t)b * NC + c0) * NN;
  const _Float16* sp =
      (br == 0 ? xh : (br == 1 ? yh : zh)) + ((size_t)b * NC + d0) * NN;
  v8f acc = {};
  for (int nk = 0; nk < NN; nk += 32) {
    v16h a = load_A_rm(xa + nk, NN);
    v16h bb = load_B_cm(sp + nk, NN);
    acc = wmma16(a, bb, acc);
  }
  int half8 = (lane < 16) ? 0 : 8, col = lane & 15;
#pragma unroll
  for (int r = 0; r < 8; ++r)
    camG[(((size_t)b * 3 + br) * 64 + c0 + r + half8) * 64 + d0 + col] = acc[r];
}

// ---------------- Kernel 5: CAM softmax (A = softmax_d(rowmin - G)) ----------
__global__ void cam_softmax(const float* __restrict__ camG, float* __restrict__ camA) {
  __shared__ float red[64];
  int row = blockIdx.x;  // B*3*64 rows
  int d = threadIdx.x;
  float g = camG[(size_t)row * 64 + d];
  red[d] = g;
  __syncthreads();
  for (int s = 32; s > 0; s >>= 1) {
    if (d < s) red[d] = fminf(red[d], red[d + s]);
    __syncthreads();
  }
  float rmin = red[0];
  __syncthreads();
  float ev = __expf(rmin - g);
  red[d] = ev;
  __syncthreads();
  for (int s = 32; s > 0; s >>= 1) {
    if (d < s) red[d] += red[d + s];
    __syncthreads();
  }
  camA[(size_t)row * 64 + d] = ev / red[0];
}

// ---------------- Kernel 6: CAM A @ sources (coalesced scalar) ---------------
__global__ void cam_av(const float* __restrict__ x, const float* __restrict__ y,
                       const float* __restrict__ z, const float* __restrict__ camA,
                       float* __restrict__ camacc) {
  int t = blockIdx.x * 256 + threadIdx.x;  // B*64*N
  int n = t & (NN - 1);
  int c = (t >> 12) & 63;
  int b = t >> 18;
  float acc = 0.f;
#pragma unroll
  for (int br = 0; br < 3; ++br) {
    const float* sp = (br == 0 ? x : (br == 1 ? y : z)) + (size_t)b * NC * NN;
    const float* Ar = camA + (((size_t)b * 3 + br) * 64 + c) * 64;
    for (int d = 0; d < 64; ++d) acc += Ar[d] * sp[(size_t)d * NN + n];
  }
  camacc[((size_t)b * NC + c) * NN + n] = acc;
}

// ---------------- Kernel 7: final 1x1 convs + residuals ----------------------
__global__ void final_kernel(const float* __restrict__ x,
                             const float* __restrict__ pamacc,
                             const float* __restrict__ camacc,
                             const float* __restrict__ pam_wl,
                             const float* __restrict__ pam_bl,
                             const float* __restrict__ pam_gamma,
                             const float* __restrict__ cam_wl,
                             const float* __restrict__ cam_bl,
                             const float* __restrict__ cam_gamma,
                             float* __restrict__ out) {
  int t = blockIdx.x * 256 + threadIdx.x;  // B*64*N
  int n = t & (NN - 1);
  int c = (t >> 12) & 63;
  int b = t >> 18;
  float ap = pam_bl[c], ac = cam_bl[c];
  const float* Pb = pamacc + (size_t)b * NC * NN;
  const float* Cb = camacc + (size_t)b * NC * NN;
  for (int d = 0; d < 64; ++d) {
    ap += pam_wl[c * 64 + d] * Pb[(size_t)d * NN + n];
    ac += cam_wl[c * 64 + d] * Cb[(size_t)d * NN + n];
  }
  float xi = x[((size_t)b * NC + c) * NN + n];
  out[(size_t)t] = 2.f * xi + pam_gamma[0] * ap + cam_gamma[0] * ac;
}

// ---------------- Launcher ---------------------------------------------------
extern "C" void kernel_launch(void* const* d_in, const int* in_sizes, int n_in,
                              void* d_out, int out_size, void* d_ws, size_t ws_size,
                              hipStream_t stream) {
  const float* x = (const float*)d_in[0];
  const float* y = (const float*)d_in[1];
  const float* z = (const float*)d_in[2];
  const float* wq = (const float*)d_in[3];
  const float* bq = (const float*)d_in[4];
  const float* wk1 = (const float*)d_in[5];
  const float* bk1 = (const float*)d_in[6];
  const float* wk2 = (const float*)d_in[7];
  const float* bk2 = (const float*)d_in[8];
  const float* wk3 = (const float*)d_in[9];
  const float* bk3 = (const float*)d_in[10];
  const float* wv = (const float*)d_in[11];
  const float* bv = (const float*)d_in[12];
  const float* pwl = (const float*)d_in[13];
  const float* pbl = (const float*)d_in[14];
  const float* pg = (const float*)d_in[15];
  const float* cwl = (const float*)d_in[16];
  const float* cbl = (const float*)d_in[17];
  const float* cg = (const float*)d_in[18];
  float* out = (float*)d_out;

  char* w = (char*)d_ws;
  _Float16* Qf = (_Float16*)w; w += (size_t)NB * NN * 32 * 2;          // 512 KB
  _Float16* Kf = (_Float16*)w; w += (size_t)3 * NB * NT * 32 * 16 * 2; // 1.5 MB
  _Float16* Vh = (_Float16*)w; w += (size_t)NB * NC * NN * 2;          // 1 MB
  _Float16* xh = (_Float16*)w; w += (size_t)NB * NC * NN * 2;
  _Float16* yh = (_Float16*)w; w += (size_t)NB * NC * NN * 2;
  _Float16* zh = (_Float16*)w; w += (size_t)NB * NC * NN * 2;
  float2* MR = (float2*)w;     w += (size_t)3 * NB * NN * sizeof(float2);
  float* pamacc = (float*)w;   w += (size_t)NB * NC * NN * 4;
  float* camG = (float*)w;     w += (size_t)NB * 3 * 64 * 64 * 4;
  float* camA = (float*)w;     w += (size_t)NB * 3 * 64 * 64 * 4;
  float* camacc = (float*)w;   w += (size_t)NB * NC * NN * 4;

  proj_kernel<<<32, 256, 0, stream>>>(x, y, z, wq, bq, wk1, bk1, wk2, bk2, wk3,
                                      bk3, wv, bv, Qf, Kf, Vh, xh, yh, zh);
  pam_pass1<<<dim3(32, NB, 3), 256, 0, stream>>>(Qf, Kf, MR);
  pam_pass2<<<256, 256, 0, stream>>>(Qf, Kf, Vh, MR, pamacc);
  cam_gram<<<12, 256, 0, stream>>>(xh, yh, zh, camG);
  cam_softmax<<<384, 64, 0, stream>>>(camG, camA);
  cam_av<<<2048, 256, 0, stream>>>(x, y, z, camA, camacc);
  final_kernel<<<2048, 256, 0, stream>>>(x, pamacc, camacc, pwl, pbl, pg, cwl,
                                         cbl, cg, out);
}
